// SpatioTemporalFGNN_86053964743111
// MI455X (gfx1250) — matrix operand order repaired and checked by reading, not compiled, not run on hardware
//
#include <hip/hip_runtime.h>
#include <hip/hip_bf16.h>
#include <math.h>

// Problem constants (match reference)
#define Bb 16
#define Ss 12
#define Nn 325
#define Ee 2600
#define CIN 32
#define Hh 64
#define HEADS 4
#define COUT 12
#define BS (Bb * Ss)           // 192
#define MTOT (BS * Nn)         // 62400 rows, multiple of 16
#define MTILES (MTOT / 16)     // 3900
#define QCOLS (HEADS * Hh)     // 256
#define KFIN (Ss * QCOLS)      // 3072
#define COUTP 16               // padded output cols for final GEMM

typedef __attribute__((ext_vector_type(16))) _Float16 v16h;
typedef __attribute__((ext_vector_type(8)))  float    v8f;

#define WMMA_F16(a, b, c) \
    __builtin_amdgcn_wmma_f32_16x16x32_f16(false, (a), false, (b), (short)0, (c), false, false)

// ---------------------------------------------------------------------------
// Kernel 0: convert weight [K, ncols_src] f32 row-major -> f16 col-major
// [ncols_dst, K], zero-padding columns ncols_src..ncols_dst-1.
// ---------------------------------------------------------------------------
__global__ void convert_weight(const float* __restrict__ W, int ncols_src,
                               int K, int ncols_dst, _Float16* __restrict__ Wt)
{
    const int idx = blockIdx.x * blockDim.x + threadIdx.x;
    if (idx >= ncols_dst * K) return;
    const int col = idx / K;
    const int k   = idx % K;
    Wt[(size_t)col * K + k] =
        (col < ncols_src) ? (_Float16)W[(size_t)k * ncols_src + col] : (_Float16)0.f;
}

// ---------------------------------------------------------------------------
// Kernel 1: build CSR (incoming edges per target node) + GCN rsqrt-degrees.
// Serial count/scatter by thread 0 keeps edge ordering deterministic.
// ---------------------------------------------------------------------------
__global__ void build_graph(const int* __restrict__ edge_row,   // sources [E]
                            const int* __restrict__ edge_col,   // targets [E]
                            int* __restrict__ off,              // [N+1]
                            int* __restrict__ sorted,           // [E]
                            float* __restrict__ dinv)           // [N]
{
    __shared__ int cnt[Nn + 1];
    __shared__ int cur[Nn + 1];
    for (int i = threadIdx.x; i <= Nn; i += blockDim.x) cnt[i] = 0;
    __syncthreads();
    if (threadIdx.x == 0) {
        for (int e = 0; e < Ee; ++e) cnt[edge_col[e]]++;
        int acc = 0;
        for (int n = 0; n < Nn; ++n) { cur[n] = acc; off[n] = acc; acc += cnt[n]; }
        off[Nn] = acc;
        for (int e = 0; e < Ee; ++e) { int c = edge_col[e]; sorted[cur[c]++] = e; }
    }
    __syncthreads();
    for (int n = threadIdx.x; n < Nn; n += blockDim.x)
        dinv[n] = rsqrtf((float)cnt[n] + 1.0f);   // +1 self loop
}

// Convert one contiguous 16-float run into an A fragment (contiguous k order)
__device__ __forceinline__ v16h load_a_frag(const float* __restrict__ p)
{
    const float4 r0 = ((const float4*)p)[0];
    const float4 r1 = ((const float4*)p)[1];
    const float4 r2 = ((const float4*)(p + 8))[0];
    const float4 r3 = ((const float4*)(p + 8))[1];
    v16h a;
    a[0]  = (_Float16)r0.x; a[1]  = (_Float16)r0.y; a[2]  = (_Float16)r0.z; a[3]  = (_Float16)r0.w;
    a[4]  = (_Float16)r1.x; a[5]  = (_Float16)r1.y; a[6]  = (_Float16)r1.z; a[7]  = (_Float16)r1.w;
    a[8]  = (_Float16)r2.x; a[9]  = (_Float16)r2.y; a[10] = (_Float16)r2.z; a[11] = (_Float16)r2.w;
    a[12] = (_Float16)r3.x; a[13] = (_Float16)r3.y; a[14] = (_Float16)r3.z; a[15] = (_Float16)r3.w;
    return a;
}

// ---------------------------------------------------------------------------
// Kernel 2: WMMA GEMM  C[M, ncols] = A[M, K] @ W[K, ncols] (+bias)
// Weights pre-converted to f16 col-major [ncols, K]; one wave computes a
// 16x64 output tile (A fragment reused across 4 WMMAs). K stepped by 32.
// ncols must be a multiple of 64 here.
// ---------------------------------------------------------------------------
__global__ void gemm_wmma(const float* __restrict__ A, int lda,
                          const _Float16* __restrict__ Wt,   // [ncols][K] col-major
                          int K, const float* __restrict__ bias,
                          float* __restrict__ C, int ldc, int mtiles)
{
    const int wave = threadIdx.x >> 5;
    const int lane = threadIdx.x & 31;
    const int tm = blockIdx.x * (blockDim.x >> 5) + wave;
    if (tm >= mtiles) return;                    // whole wave exits together

    const int half = lane >> 4;
    const int lo   = lane & 15;
    const int mrow = tm * 16 + lo;
    const int colbase = blockIdx.y * 64;

    v8f acc0 = {}, acc1 = {}, acc2 = {}, acc3 = {};
    for (int k0 = 0; k0 < K; k0 += 32) {
        const v16h a = load_a_frag(A + (size_t)mrow * lda + k0 + half * 16);
        const _Float16* wb = Wt + (size_t)(colbase + lo) * K + k0 + half * 16;
        const v16h b0 = *(const v16h*)(wb);
        const v16h b1 = *(const v16h*)(wb + (size_t)16 * K);
        const v16h b2 = *(const v16h*)(wb + (size_t)32 * K);
        const v16h b3 = *(const v16h*)(wb + (size_t)48 * K);
        acc0 = WMMA_F16(a, b0, acc0);
        acc1 = WMMA_F16(a, b1, acc1);
        acc2 = WMMA_F16(a, b2, acc2);
        acc3 = WMMA_F16(a, b3, acc3);
    }

    const float bv0 = bias ? bias[colbase + lo]      : 0.0f;
    const float bv1 = bias ? bias[colbase + 16 + lo] : 0.0f;
    const float bv2 = bias ? bias[colbase + 32 + lo] : 0.0f;
    const float bv3 = bias ? bias[colbase + 48 + lo] : 0.0f;
#pragma unroll
    for (int v = 0; v < 8; ++v) {
        float* crow = C + (size_t)(tm * 16 + v + half * 8) * ldc + colbase + lo;
        crow[0]  = acc0[v] + bv0;
        crow[16] = acc1[v] + bv1;
        crow[32] = acc2[v] + bv2;
        crow[48] = acc3[v] + bv3;
    }
}

// ---------------------------------------------------------------------------
// Kernel 3: GCN normalize/scatter/ReLU. One 64-thread block per (bs, node).
// ---------------------------------------------------------------------------
__global__ void gcn_aggregate(const float* __restrict__ hw,      // [BS*N, 64]
                              const int* __restrict__ off,
                              const int* __restrict__ sorted,
                              const int* __restrict__ edge_row,
                              const float* __restrict__ dinv,
                              const float* __restrict__ bias,    // [64]
                              float* __restrict__ hout)          // [BS*N, 64]
{
    const int n  = blockIdx.x % Nn;
    const int bs = blockIdx.x / Nn;
    const int f  = threadIdx.x;
    const float dn = dinv[n];
    const float* base = hw + (size_t)bs * Nn * Hh;
    float acc = base[(size_t)n * Hh + f] * dn * dn;      // self loop
    const int s0 = off[n], s1 = off[n + 1];
    for (int i = s0; i < s1; ++i) {
        const int src = edge_row[sorted[i]];
        acc += base[(size_t)src * Hh + f] * dinv[src] * dn;
    }
    hout[((size_t)bs * Nn + n) * Hh + f] = fmaxf(acc + bias[f], 0.0f);
}

// ---------------------------------------------------------------------------
// Kernel 4: edge attention + segment softmax + aggregate + root skip.
// One wave32 per (bs, node, head); each lane owns 2 of 64 features.
// Online softmax over incoming edges. Writes out = attn + skip (in place OK).
// ---------------------------------------------------------------------------
__global__ void attention(const float* __restrict__ q,     // [BS*N, 256]
                          const float* __restrict__ k,
                          const float* __restrict__ v,
                          const float* __restrict__ skip,   // xt@Ws + bs
                          const int* __restrict__ off,
                          const int* __restrict__ sorted,
                          const int* __restrict__ edge_row,
                          float* __restrict__ outp)         // [BS*N, 256]
{
    const int wave = threadIdx.x >> 5;
    const int lane = threadIdx.x & 31;
    const int unit = blockIdx.x * (blockDim.x >> 5) + wave;
    if (unit >= BS * Nn * HEADS) return;
    const int head = unit & (HEADS - 1);
    const int rest = unit >> 2;
    const int n    = rest % Nn;
    const int bs   = rest / Nn;

    const size_t vecbase = (((size_t)bs * Nn + n) * HEADS + head) * Hh;
    const float q0 = q[vecbase + 2 * lane];
    const float q1 = q[vecbase + 2 * lane + 1];

    float m = -INFINITY, l = 0.0f, a0 = 0.0f, a1 = 0.0f;
    const int s0 = off[n], s1 = off[n + 1];
    for (int i = s0; i < s1; ++i) {
        const int src = edge_row[sorted[i]];
        const size_t sb = (((size_t)bs * Nn + src) * HEADS + head) * Hh;
        float p = q0 * k[sb + 2 * lane] + q1 * k[sb + 2 * lane + 1];
#pragma unroll
        for (int o = 16; o > 0; o >>= 1) p += __shfl_xor(p, o, 32);
        const float logit = p * 0.125f;                 // 1/sqrt(H)
        const float mnew  = fmaxf(m, logit);
        const float scale = __expf(m - mnew);           // exp(-inf)=0 first iter
        const float w     = __expf(logit - mnew);
        l  = l * scale + w;
        a0 = a0 * scale + w * v[sb + 2 * lane];
        a1 = a1 * scale + w * v[sb + 2 * lane + 1];
        m  = mnew;
    }
    const float denom = l + 1e-16f;
    outp[vecbase + 2 * lane]     = a0 / denom + skip[vecbase + 2 * lane];
    outp[vecbase + 2 * lane + 1] = a1 / denom + skip[vecbase + 2 * lane + 1];
}

// ---------------------------------------------------------------------------
// Kernel 5: final projection y[B*N, 12] = gathered out[B,S,N,256] @ Wl + bl.
// Row = (b, n), K index = (s, f). Wl pre-padded/transposed to f16 [16][3072].
// A runs never cross an s boundary since k0 % 32 == 0.
// ---------------------------------------------------------------------------
__global__ void final_gemm(const float* __restrict__ outbuf,     // [B,S,N,256]
                           const _Float16* __restrict__ Wlt,     // [16][3072]
                           const float* __restrict__ bl,         // [12]
                           float* __restrict__ y)                // [B*N, 12]
{
    const int wave = threadIdx.x >> 5;
    const int lane = threadIdx.x & 31;
    const int tile = blockIdx.x * (blockDim.x >> 5) + wave;
    if (tile >= (Bb * Nn) / 16) return;        // 325 tiles exactly

    const int half = lane >> 4;
    const int lo   = lane & 15;
    const int mrow = tile * 16 + lo;
    const int bidx = mrow / Nn;
    const int node = mrow % Nn;

    v8f acc = {};
    for (int k0 = 0; k0 < KFIN; k0 += 32) {
        const int s1 = k0 >> 8;
        const int f1 = k0 & 255;
        const float* abase =
            outbuf + (((size_t)(bidx * Ss + s1)) * Nn + node) * QCOLS + f1 + half * 16;
        // one 16-float contiguous run -- same layout as generic A fragment
        const v16h a = load_a_frag(abase);
        const v16h b = *(const v16h*)(Wlt + (size_t)lo * KFIN + k0 + half * 16);
        acc = WMMA_F16(a, b, acc);
    }
    if (lo < COUT) {
        const float bv = bl[lo];
#pragma unroll
        for (int v = 0; v < 8; ++v) {
            const int row = tile * 16 + v + half * 8;
            y[(size_t)row * COUT + lo] = acc[v] + bv;
        }
    }
}

// ---------------------------------------------------------------------------
// Host-side launcher
// ---------------------------------------------------------------------------
extern "C" void kernel_launch(void* const* d_in, const int* in_sizes, int n_in,
                              void* d_out, int out_size, void* d_ws, size_t ws_size,
                              hipStream_t stream) {
    (void)in_sizes; (void)n_in; (void)out_size; (void)ws_size;
    const float* x    = (const float*)d_in[0];
    const int*   ei   = (const int*)d_in[1];
    const float* Wg[4] = { (const float*)d_in[2], (const float*)d_in[4],
                           (const float*)d_in[6], (const float*)d_in[8] };
    const float* bg[4] = { (const float*)d_in[3], (const float*)d_in[5],
                           (const float*)d_in[7], (const float*)d_in[9] };
    const float* Wq = (const float*)d_in[10]; const float* bq = (const float*)d_in[11];
    const float* Wk = (const float*)d_in[12]; const float* bk = (const float*)d_in[13];
    const float* Wv = (const float*)d_in[14]; const float* bv = (const float*)d_in[15];
    const float* Wsk = (const float*)d_in[16]; const float* bsk = (const float*)d_in[17];
    const float* Wl = (const float*)d_in[18]; const float* bl = (const float*)d_in[19];
    float* y = (float*)d_out;

    const int* erow = ei;        // edge_index[0] = sources
    const int* ecol = ei + Ee;   // edge_index[1] = targets

    // Workspace carve-up (256B aligned)
    char* wsp = (char*)d_ws;
    size_t pos = 0;
    auto take = [&](size_t bytes) -> char* {
        char* p = wsp + pos;
        pos += (bytes + 255) & ~(size_t)255;
        return p;
    };
    int*   off    = (int*)  take((Nn + 1) * sizeof(int));
    int*   sorted = (int*)  take(Ee * sizeof(int));
    float* dinv   = (float*)take(Nn * sizeof(float));
    _Float16* Wg0t = (_Float16*)take((size_t)Hh * CIN * sizeof(_Float16));
    _Float16* Wgt[4] = { Wg0t,
                         (_Float16*)take((size_t)Hh * Hh * sizeof(_Float16)),
                         (_Float16*)take((size_t)Hh * Hh * sizeof(_Float16)),
                         (_Float16*)take((size_t)Hh * Hh * sizeof(_Float16)) };
    _Float16* Wqt = (_Float16*)take((size_t)QCOLS * Hh * sizeof(_Float16));
    _Float16* Wkt = (_Float16*)take((size_t)QCOLS * Hh * sizeof(_Float16));
    _Float16* Wvt = (_Float16*)take((size_t)QCOLS * Hh * sizeof(_Float16));
    _Float16* Wst = (_Float16*)take((size_t)QCOLS * Hh * sizeof(_Float16));
    _Float16* Wlt = (_Float16*)take((size_t)COUTP * KFIN * sizeof(_Float16));
    float* h0     = (float*)take((size_t)MTOT * Hh * sizeof(float));
    float* h1     = (float*)take((size_t)MTOT * Hh * sizeof(float));
    float* qb     = (float*)take((size_t)MTOT * QCOLS * sizeof(float));
    float* kb     = (float*)take((size_t)MTOT * QCOLS * sizeof(float));
    float* vb     = (float*)take((size_t)MTOT * QCOLS * sizeof(float));
    float* sk     = (float*)take((size_t)MTOT * QCOLS * sizeof(float));
    float* hw     = qb;   // GEMM scratch aliases qb (qb written afterwards)

    // 0. Weight conversion to f16 col-major (padded where needed)
    auto conv = [&](const float* W, int ncs, int K, int ncd, _Float16* Wt) {
        const int tot = ncd * K;
        convert_weight<<<(tot + 255) / 256, 256, 0, stream>>>(W, ncs, K, ncd, Wt);
    };
    conv(Wg[0], Hh, CIN, Hh, Wgt[0]);
    conv(Wg[1], Hh, Hh, Hh, Wgt[1]);
    conv(Wg[2], Hh, Hh, Hh, Wgt[2]);
    conv(Wg[3], Hh, Hh, Hh, Wgt[3]);
    conv(Wq,  QCOLS, Hh, QCOLS, Wqt);
    conv(Wk,  QCOLS, Hh, QCOLS, Wkt);
    conv(Wv,  QCOLS, Hh, QCOLS, Wvt);
    conv(Wsk, QCOLS, Hh, QCOLS, Wst);
    conv(Wl,  COUT, KFIN, COUTP, Wlt);

    // 1. Graph structure
    build_graph<<<1, 256, 0, stream>>>(erow, ecol, off, sorted, dinv);

    // 2. Four GCN layers (GEMM -> normalize/scatter/ReLU), ping-pong h0/h1
    const int WPB = 8;                               // waves per block
    const dim3 blk(32 * WPB);
    {
        const dim3 g0((MTILES + WPB - 1) / WPB, Hh / 64);   // 64-col tile
        // layer 0: x [MTOT,32] @ Wg0 [32,64]
        gemm_wmma<<<g0, blk, 0, stream>>>(x, CIN, Wgt[0], CIN, nullptr, hw, Hh, MTILES);
        gcn_aggregate<<<MTOT, Hh, 0, stream>>>(hw, off, sorted, erow, dinv, bg[0], h0);
        const float* hin = h0; float* hout = h1;
        for (int layer = 1; layer < 4; ++layer) {
            gemm_wmma<<<g0, blk, 0, stream>>>(hin, Hh, Wgt[layer], Hh, nullptr, hw, Hh, MTILES);
            gcn_aggregate<<<MTOT, Hh, 0, stream>>>(hw, off, sorted, erow, dinv, bg[layer], hout);
            const float* t = hin; hin = hout; hout = (float*)t;
        }
    }
    const float* hfin = h1;   // after layers 0(h0),1(h1),2(h0),3(h1)

    // 3. Q/K/V/skip projections [MTOT,64] @ [64,256] + bias
    {
        const dim3 gq((MTILES + WPB - 1) / WPB, QCOLS / 64);
        gemm_wmma<<<gq, blk, 0, stream>>>(hfin, Hh, Wqt, Hh, bq,  qb, QCOLS, MTILES);
        gemm_wmma<<<gq, blk, 0, stream>>>(hfin, Hh, Wkt, Hh, bk,  kb, QCOLS, MTILES);
        gemm_wmma<<<gq, blk, 0, stream>>>(hfin, Hh, Wvt, Hh, bv,  vb, QCOLS, MTILES);
        gemm_wmma<<<gq, blk, 0, stream>>>(hfin, Hh, Wst, Hh, bsk, sk, QCOLS, MTILES);
    }

    // 4. Attention: one wave per (bs, node, head); out = attn + skip (in place)
    {
        const int units  = BS * Nn * HEADS;          // 249600
        const int blocks = (units + WPB - 1) / WPB;  // 31200
        attention<<<blocks, blk, 0, stream>>>(qb, kb, vb, sk, off, sorted, erow, sk);
    }

    // 5. Final projection with gathered A and padded f16 weights
    {
        const int ftiles = (Bb * Nn) / 16;           // 325
        final_gemm<<<(ftiles + WPB - 1) / WPB, blk, 0, stream>>>(sk, Wlt, bl, y);
    }
}
